// S_Mamba_26250840113634
// MI455X (gfx1250) — compile-verified, matrix operand order checked by reading, not run on hardware
//
#include <hip/hip_runtime.h>
#include <hip/hip_bf16.h>
#include <math.h>

// ---------------- problem constants ----------------
#define BD     8
#define TT     323          // tokens per batch = N_VARS + 2
#define LH     96
#define NVARS  321
#define DM     512
#define DST    16
#define PRED   336
#define MROWS  (BD * TT)    // 2584

#define CDIV(a,b) (((a)+(b)-1)/(b))

typedef __attribute__((ext_vector_type(16))) __bf16 v16bf;
typedef __attribute__((ext_vector_type(8)))  __bf16 v8bf;
typedef __attribute__((ext_vector_type(8)))  float  v8f;

union BFrag { v16bf v; v8bf h[2]; };

__device__ __forceinline__ unsigned short f2bf(float f) {
  union { float f; unsigned u; } c; c.f = f;
  unsigned u = c.u;
  u += 0x7FFFu + ((u >> 16) & 1u);   // round-to-nearest-even truncation
  return (unsigned short)(u >> 16);
}

// -------- f32 -> bf16 bulk conversion (8 elems / thread, 16B stores) --------
__global__ void cvt_bf16_kernel(const float* __restrict__ src, __bf16* __restrict__ dst, int n8) {
  int i = blockIdx.x * blockDim.x + threadIdx.x;
  if (i >= n8) return;
  const float4* s = reinterpret_cast<const float4*>(src) + (size_t)i * 2;
  float4 x = s[0], y = s[1];
  union { v8bf v; unsigned short u[8]; } o;
  o.u[0] = f2bf(x.x); o.u[1] = f2bf(x.y); o.u[2] = f2bf(x.z); o.u[3] = f2bf(x.w);
  o.u[4] = f2bf(y.x); o.u[5] = f2bf(y.y); o.u[6] = f2bf(y.z); o.u[7] = f2bf(y.w);
  reinterpret_cast<v8bf*>(dst)[i] = o.v;
}

__device__ __forceinline__ v8f wmma_bf(v16bf a, v16bf b, v8f c) {
  return __builtin_amdgcn_wmma_f32_16x16x32_bf16(false, a, false, b, (short)0, c, false, false);
}

__device__ __forceinline__ void load_frag(const __bf16* __restrict__ p, BFrag& fr) {
  // elements 0..7 = K[k0+sel*8 .. +7], 8..15 = K[k0+16+sel*8 .. +7]
  fr.h[0] = *reinterpret_cast<const v8bf*>(p);
  fr.h[1] = *reinterpret_cast<const v8bf*>(p + 16);
}

__device__ __forceinline__ void store_tile(v8f acc, float* __restrict__ C, int ldc,
                                           int mbase, int n, int M,
                                           const float* __restrict__ bias, int act) {
  float bv = bias ? bias[n] : 0.f;
#pragma unroll
  for (int r = 0; r < 8; ++r) {
    int m = mbase + r;
    if (m < M) {
      float v = acc[r] + bv;
      if (act == 1)      v = (v > 20.f) ? v : log1pf(__expf(v));                 // softplus
      else if (act == 2) v = 0.5f * v * (1.f + erff(v * 0.70710678118654752f));  // exact gelu
      C[(size_t)m * ldc + n] = v;
    }
  }
}

// Generic GEMM: C[M,N] = act(A[M,K](lda,bf16) * W[N,K]^T(bf16) + bias)
// block = (32,4): 4 waves, each wave owns a 32(M) x NV*16(N) strip, branch-free
// inner loop (NV is a compile-time subtile count; n-range exact by construction).
template <int NV>
__global__ void __launch_bounds__(128)
gemm_bf16_wmma(const __bf16* __restrict__ A, int lda,
               const __bf16* __restrict__ W,
               const float* __restrict__ bias,
               float* __restrict__ C, int ldc,
               int M, int K, int act, int n0base) {
  const int lane = threadIdx.x;
  const int wv   = threadIdx.y;
  const int m0   = blockIdx.y * 128 + wv * 32;
  if (m0 >= M) return;                       // whole-wave uniform exit
  const int nbase = n0base + blockIdx.x * 64;
  const int l16 = lane & 15;
  const int sel = lane >> 4;                 // which K-half this lane holds

  int ar0 = m0 + l16;      if (ar0 >= M) ar0 = M - 1;   // clamp (stores guarded)
  int ar1 = m0 + 16 + l16; if (ar1 >= M) ar1 = M - 1;
  const __bf16* a0 = A + (size_t)ar0 * lda + sel * 8;
  const __bf16* a1 = A + (size_t)ar1 * lda + sel * 8;

  const __bf16* wp[NV];
  v8f acc0[NV], acc1[NV];
  v8f zv = {0,0,0,0,0,0,0,0};
#pragma unroll
  for (int j = 0; j < NV; ++j) {
    wp[j] = W + (size_t)(nbase + j * 16 + l16) * K + sel * 8;
    acc0[j] = zv; acc1[j] = zv;
  }

  for (int k0 = 0; k0 < K; k0 += 32) {
    BFrag fa0, fa1;
    load_frag(a0 + k0, fa0);
    load_frag(a1 + k0, fa1);
#pragma unroll
    for (int j = 0; j < NV; ++j) {
      BFrag fb;
      load_frag(wp[j] + k0, fb);
      acc0[j] = wmma_bf(fa0.v, fb.v, acc0[j]);
      acc1[j] = wmma_bf(fa1.v, fb.v, acc1[j]);
    }
  }

  // C/D layout: VGPR r -> M = r + 8*sel (within 16-row tile), N = lane%16
  const int mb0 = m0 + sel * 8;
  const int mb1 = m0 + 16 + sel * 8;
#pragma unroll
  for (int j = 0; j < NV; ++j) {
    store_tile(acc0[j], C, ldc, mb0, nbase + j * 16 + l16, M, bias, act);
    store_tile(acc1[j], C, ldc, mb1, nbase + j * 16 + l16, M, bias, act);
  }
}

// -------- per-(b,var) mean / stdev over the 96 history steps --------
__global__ void meanstd_kernel(const float* __restrict__ hist, float* __restrict__ ms) {
  int idx = blockIdx.x * blockDim.x + threadIdx.x;
  if (idx >= BD * NVARS) return;
  int b = idx / NVARS, v = idx % NVARS;
  float s = 0.f;
  for (int t = 0; t < LH; ++t) s += hist[((size_t)(b * LH + t) * NVARS + v) * 3];
  float mean = s / (float)LH;
  float q = 0.f;
  for (int t = 0; t < LH; ++t) {
    float d = hist[((size_t)(b * LH + t) * NVARS + v) * 3] - mean;
    q += d * d;
  }
  ms[idx * 2]     = mean;
  ms[idx * 2 + 1] = sqrtf(q / (float)LH + 1e-5f);
}

// -------- tokens (B, 323, 96): normalized vars + 2 time-mark rows --------
__global__ void tokens_kernel(const float* __restrict__ hist, const float* __restrict__ ms,
                              float* __restrict__ tok) {
  int idx = blockIdx.x * blockDim.x + threadIdx.x;
  if (idx >= BD * TT * LH) return;
  int t  = idx % LH;
  int rv = (idx / LH) % TT;
  int b  = idx / (LH * TT);
  float val;
  if (rv < NVARS) {
    float x = hist[((size_t)(b * LH + t) * NVARS + rv) * 3];
    const float* m = ms + (b * NVARS + rv) * 2;
    val = (x - m[0]) / m[1];
  } else if (rv == NVARS) {
    float c1 = hist[((size_t)(b * LH + t) * NVARS + 0) * 3 + 1];
    val = floorf(c1 * 24.f) / 23.f;                     // tod transform
  } else {
    val = hist[((size_t)(b * LH + t) * NVARS + 0) * 3 + 2];
  }
  tok[idx] = val;
}

// -------- depthwise conv (width 2) + SiLU, direction aware --------
__global__ void conv_silu_kernel(const float* __restrict__ xz, const float* __restrict__ cw,
                                 const float* __restrict__ cb, float* __restrict__ out, int rev) {
  int idx = blockIdx.x * blockDim.x + threadIdx.x;
  if (idx >= MROWS * DM) return;
  int d = idx & (DM - 1);
  int r = idx >> 9;
  int t = r % TT;
  float xc = xz[(size_t)r * 1024 + d];
  float xp = 0.f;
  if (rev) { if (t + 1 < TT) xp = xz[(size_t)(r + 1) * 1024 + d]; }
  else     { if (t > 0)      xp = xz[(size_t)(r - 1) * 1024 + d]; }
  float v = xp * cw[d * 2] + xc * cw[d * 2 + 1] + cb[d];
  out[(size_t)r * DM + d] = v / (1.f + __expf(-v));     // silu
}

// -------- selective scan: one thread per (b, d); 16 states in registers --------
__global__ void scan_kernel(const float* __restrict__ dt, const float* __restrict__ xi,
                            const float* __restrict__ dbc, const float* __restrict__ xz,
                            const float* __restrict__ Alog, const float* __restrict__ Dp,
                            float* __restrict__ yout, int rev) {
  int idx = blockIdx.x * blockDim.x + threadIdx.x;
  if (idx >= BD * DM) return;
  int d = idx & (DM - 1);
  int b = idx >> 9;
  float Av[DST];
#pragma unroll
  for (int s = 0; s < DST; ++s) Av[s] = -__expf(Alog[d * DST + s]);
  float Dd = Dp[d];
  float h[DST];
#pragma unroll
  for (int s = 0; s < DST; ++s) h[s] = 0.f;

  for (int i = 0; i < TT; ++i) {
    int t = rev ? (TT - 1 - i) : i;
    size_t r = (size_t)b * TT + t;
    if (i + 1 < TT) {                                   // prefetch next step
      int tn = rev ? (TT - 2 - i) : (i + 1);
      size_t rn = (size_t)b * TT + tn;
      __builtin_prefetch(dt  + rn * DM + d, 0, 1);
      __builtin_prefetch(xi  + rn * DM + d, 0, 1);
      __builtin_prefetch(dbc + rn * 64, 0, 1);
    }
    float dtv = dt[r * DM + d];
    float xv  = xi[r * DM + d];
    const float* bc = dbc + r * 64;
    float y = 0.f;
#pragma unroll
    for (int s = 0; s < DST; ++s) {
      h[s] = __expf(dtv * Av[s]) * h[s] + dtv * bc[32 + s] * xv;
      y += h[s] * bc[48 + s];
    }
    float z = xz[r * 1024 + DM + d];
    yout[r * DM + d] = (y + xv * Dd) * (z / (1.f + __expf(-z)));  // * silu(z)
  }
}

// -------- layernorm over D=512 with up to 3-way residual sum fused --------
__global__ void ln_kernel(const float* __restrict__ a, const float* __restrict__ b,
                          const float* __restrict__ c, const float* __restrict__ g,
                          const float* __restrict__ beta, float* __restrict__ out) {
  int row = blockIdx.x;
  int tid = threadIdx.x;              // 256 threads, 2 elems each
  size_t base = (size_t)row * DM;
  int i0 = tid * 2, i1 = i0 + 1;
  float x0 = a[base + i0], x1 = a[base + i1];
  if (b) { x0 += b[base + i0]; x1 += b[base + i1]; }
  if (c) { x0 += c[base + i0]; x1 += c[base + i1]; }
  __shared__ float red[256];
  red[tid] = x0 + x1;
  __syncthreads();
  for (int s = 128; s > 0; s >>= 1) { if (tid < s) red[tid] += red[tid + s]; __syncthreads(); }
  float mean = red[0] / (float)DM;
  __syncthreads();
  float d0 = x0 - mean, d1 = x1 - mean;
  red[tid] = d0 * d0 + d1 * d1;
  __syncthreads();
  for (int s = 128; s > 0; s >>= 1) { if (tid < s) red[tid] += red[tid + s]; __syncthreads(); }
  float inv = rsqrtf(red[0] / (float)DM + 1e-5f);
  out[base + i0] = d0 * inv * g[i0] + beta[i0];
  out[base + i1] = d1 * inv * g[i1] + beta[i1];
}

// -------- de-normalize + transpose to (B, PRED, NVARS, 1) --------
__global__ void out_kernel(const float* __restrict__ dec, const float* __restrict__ ms,
                           float* __restrict__ out) {
  int idx = blockIdx.x * blockDim.x + threadIdx.x;
  if (idx >= BD * PRED * NVARS) return;
  int v = idx % NVARS;
  int p = (idx / NVARS) % PRED;
  int b = idx / (NVARS * PRED);
  const float* m = ms + (b * NVARS + v) * 2;
  out[idx] = dec[((size_t)b * TT + v) * PRED + p] * m[1] + m[0];
}

// ---------------- host side ----------------
static inline void cvt(hipStream_t s, const float* src, __bf16* dst, size_t n) {
  int n8 = (int)(n / 8);
  cvt_bf16_kernel<<<CDIV(n8, 256), 256, 0, s>>>(src, dst, n8);
}

static inline void gemm(hipStream_t s, const __bf16* A, int lda, const __bf16* W,
                        const float* bias, float* C, int ldc, int M, int N, int K, int act) {
  dim3 b(32, 4);
  int gy = CDIV(M, 128);
  int nfull = N / 64;                 // full 64-wide column blocks -> NV=4
  if (nfull > 0) {
    dim3 g(nfull, gy);
    gemm_bf16_wmma<4><<<g, b, 0, s>>>(A, lda, W, bias, C, ldc, M, K, act, 0);
  }
  int rem = (N - nfull * 64) / 16;    // tail subtiles (N always a multiple of 16)
  if (rem > 0) {
    dim3 g(1, gy);
    if (rem == 1)      gemm_bf16_wmma<1><<<g, b, 0, s>>>(A, lda, W, bias, C, ldc, M, K, act, nfull * 64);
    else if (rem == 2) gemm_bf16_wmma<2><<<g, b, 0, s>>>(A, lda, W, bias, C, ldc, M, K, act, nfull * 64);
    else               gemm_bf16_wmma<3><<<g, b, 0, s>>>(A, lda, W, bias, C, ldc, M, K, act, nfull * 64);
  }
}

extern "C" void kernel_launch(void* const* d_in, const int* in_sizes, int n_in,
                              void* d_out, int out_size, void* d_ws, size_t ws_size,
                              hipStream_t stream) {
  const float* hist   = (const float*)d_in[0];
  const float* W_emb  = (const float*)d_in[5];
  const float* b_emb  = (const float*)d_in[6];
  const float* m_in   = (const float*)d_in[7];
  const float* m_cw   = (const float*)d_in[8];
  const float* m_cb   = (const float*)d_in[9];
  const float* m_xw   = (const float*)d_in[10];
  const float* m_dtw  = (const float*)d_in[11];
  const float* m_dtb  = (const float*)d_in[12];
  const float* m_Alog = (const float*)d_in[13];
  const float* m_D    = (const float*)d_in[14];
  const float* m_ow   = (const float*)d_in[15];
  const float* w1     = (const float*)d_in[16];
  const float* b1     = (const float*)d_in[17];
  const float* w2     = (const float*)d_in[18];
  const float* b2     = (const float*)d_in[19];
  const float* n1g    = (const float*)d_in[20];
  const float* n1b    = (const float*)d_in[21];
  const float* n2g    = (const float*)d_in[22];
  const float* n2b    = (const float*)d_in[23];
  const float* nfg    = (const float*)d_in[24];
  const float* nfb    = (const float*)d_in[25];
  const float* projW  = (const float*)d_in[26];
  const float* projb  = (const float*)d_in[27];
  float* out = (float*)d_out;
  float* ws  = (float*)d_ws;

  // ---- f32 workspace layout (elements, all 16-aligned) ----
  size_t OFF_TOK = 0;                                  // 2584*96
  size_t OFF_MS  = OFF_TOK + (size_t)MROWS * LH;       // 8*321*2
  size_t OFF_ENC = OFF_MS  + (size_t)BD * NVARS * 2;
  size_t OFF_XZ  = OFF_ENC + (size_t)MROWS * DM;       // 2584*1024 (reused: ffn2, dec)
  size_t OFF_XIC = OFF_XZ  + (size_t)MROWS * 1024;
  size_t OFF_DBC = OFF_XIC + (size_t)MROWS * DM;       // 2584*64
  size_t OFF_DT  = OFF_DBC + (size_t)MROWS * 64;
  size_t OFF_YG  = OFF_DT  + (size_t)MROWS * DM;       // reused: final-LN out
  size_t OFF_FWD = OFF_YG  + (size_t)MROWS * DM;
  size_t OFF_REV = OFF_FWD + (size_t)MROWS * DM;
  size_t OFF_X   = OFF_REV + (size_t)MROWS * DM;
  size_t OFF_F1  = OFF_X   + (size_t)MROWS * DM;
  size_t F32_END = OFF_F1  + (size_t)MROWS * DM;       // 13,648,656 floats

  float* tok  = ws + OFF_TOK;
  float* ms   = ws + OFF_MS;
  float* enc  = ws + OFF_ENC;
  float* xz   = ws + OFF_XZ;
  float* xic  = ws + OFF_XIC;
  float* dbc  = ws + OFF_DBC;
  float* dtb  = ws + OFF_DT;
  float* yg   = ws + OFF_YG;
  float* fwdb = ws + OFF_FWD;
  float* revb = ws + OFF_REV;
  float* xb   = ws + OFF_X;
  float* f1   = ws + OFF_F1;

  // ---- bf16 workspace (after f32 region): persistent weights + A-strip ----
  __bf16* bfb   = (__bf16*)(ws + F32_END);
  __bf16* wbEmb = bfb;                                  // 512*96
  __bf16* wbIn  = wbEmb + (size_t)DM * LH;              // 4*1024*512
  __bf16* wbXw  = wbIn  + (size_t)4 * 1024 * DM;        // 4*64*512
  __bf16* wbDtw = wbXw  + (size_t)4 * 64 * DM;          // 4*512*32
  __bf16* wbOw  = wbDtw + (size_t)4 * DM * 32;          // 4*512*512
  __bf16* wbW1  = wbOw  + (size_t)4 * DM * DM;          // 2*512*512
  __bf16* wbW2  = wbW1  + (size_t)2 * DM * DM;          // 2*512*512
  __bf16* wbPj  = wbW2  + (size_t)2 * DM * DM;          // 336*512
  __bf16* abf   = wbPj  + (size_t)PRED * DM;            // A strip: MROWS*DM max

  // 0. weights -> bf16 (once per launch; deterministic)
  cvt(stream, W_emb, wbEmb, (size_t)DM * LH);
  cvt(stream, m_in,  wbIn,  (size_t)4 * 1024 * DM);
  cvt(stream, m_xw,  wbXw,  (size_t)4 * 64 * DM);
  cvt(stream, m_dtw, wbDtw, (size_t)4 * DM * 32);
  cvt(stream, m_ow,  wbOw,  (size_t)4 * DM * DM);
  cvt(stream, w1,    wbW1,  (size_t)2 * DM * DM);
  cvt(stream, w2,    wbW2,  (size_t)2 * DM * DM);
  cvt(stream, projW, wbPj,  (size_t)PRED * DM);

  // 1. stats + tokens
  meanstd_kernel<<<CDIV(BD * NVARS, 256), 256, 0, stream>>>(hist, ms);
  tokens_kernel<<<CDIV(BD * TT * LH, 256), 256, 0, stream>>>(hist, ms, tok);

  // 2. embedding: enc = tokens @ W_emb^T + b_emb
  cvt(stream, tok, abf, (size_t)MROWS * LH);
  gemm(stream, abf, LH, wbEmb, b_emb, enc, DM, MROWS, DM, LH, 0);

  // 3. encoder layers
  for (int l = 0; l < 2; ++l) {
    for (int dir = 0; dir < 2; ++dir) {
      int li = l * 2 + dir;
      const float* cw  = m_cw   + (size_t)li * DM * 2;
      const float* cb  = m_cb   + (size_t)li * DM;
      const float* dtB = m_dtb  + (size_t)li * DM;
      const float* Al  = m_Alog + (size_t)li * DM * DST;
      const float* Dp  = m_D    + (size_t)li * DM;

      cvt(stream, enc, abf, (size_t)MROWS * DM);
      gemm(stream, abf, DM, wbIn + (size_t)li * 1024 * DM, nullptr,
           xz, 1024, MROWS, 1024, DM, 0);                                         // in_proj
      conv_silu_kernel<<<CDIV(MROWS * DM, 256), 256, 0, stream>>>(xz, cw, cb, xic, dir);
      cvt(stream, xic, abf, (size_t)MROWS * DM);
      gemm(stream, abf, DM, wbXw + (size_t)li * 64 * DM, nullptr,
           dbc, 64, MROWS, 64, DM, 0);                                            // x_proj
      cvt(stream, dbc, abf, (size_t)MROWS * 64);
      gemm(stream, abf, 64, wbDtw + (size_t)li * DM * 32, dtB,
           dtb, DM, MROWS, DM, 32, 1);                                            // dt + softplus
      scan_kernel<<<CDIV(BD * DM, 256), 256, 0, stream>>>(dtb, xic, dbc, xz, Al, Dp, yg, dir);
      cvt(stream, yg, abf, (size_t)MROWS * DM);
      gemm(stream, abf, DM, wbOw + (size_t)li * DM * DM, nullptr,
           dir ? revb : fwdb, DM, MROWS, DM, DM, 0);                              // out_proj
    }
    ln_kernel<<<MROWS, 256, 0, stream>>>(enc, fwdb, revb, n1g + l * DM, n1b + l * DM, xb);
    cvt(stream, xb, abf, (size_t)MROWS * DM);
    gemm(stream, abf, DM, wbW1 + (size_t)l * DM * DM, b1 + l * DM,
         f1, DM, MROWS, DM, DM, 2);                                               // ffn1 + gelu
    cvt(stream, f1, abf, (size_t)MROWS * DM);
    gemm(stream, abf, DM, wbW2 + (size_t)l * DM * DM, b2 + l * DM,
         xz, DM, MROWS, DM, DM, 0);                                               // ffn2
    ln_kernel<<<MROWS, 256, 0, stream>>>(xb, xz, nullptr, n2g + l * DM, n2b + l * DM, enc);
  }

  // 4. final norm + projection + de-norm
  ln_kernel<<<MROWS, 256, 0, stream>>>(enc, nullptr, nullptr, nfg, nfb, yg);
  cvt(stream, yg, abf, (size_t)MROWS * DM);
  gemm(stream, abf, DM, wbPj, projb, xz, PRED, MROWS, PRED, DM, 0);   // dec into xz scratch
  out_kernel<<<CDIV(BD * PRED * NVARS, 256), 256, 0, stream>>>(xz, ms, out);
}
